// MultiheadSelfAttention_72026601554197
// MI455X (gfx1250) — matrix operand order repaired and checked by reading, not compiled, not run on hardware
//
#include <hip/hip_runtime.h>
#include <hip/hip_bf16.h>
#include <stdint.h>

#define HIDDEN  1024
#define HEADS   16
#define HEADDIM 64
#define SEQ     2048
#define BATCH   4
#define L2E     1.4426950408889634f

typedef __bf16    bf16x16 __attribute__((ext_vector_type(16)));
typedef float     f32x8   __attribute__((ext_vector_type(8)));
typedef uint32_t  u32x4   __attribute__((ext_vector_type(4)));
typedef int       i32x4   __attribute__((ext_vector_type(4)));
typedef int       i32x8   __attribute__((ext_vector_type(8)));

__device__ __forceinline__ bf16x16 frag_from(u32x4 lo, u32x4 hi) {
  union { struct { u32x4 lo, hi; } u; bf16x16 f; } cv;
  cv.u.lo = lo; cv.u.hi = hi;
  return cv.f;
}

__device__ __forceinline__ uint16_t f32_to_bf16_rn(float f) {
  union { float f; uint32_t u; } v; v.f = f;
  uint32_t u = v.u + 0x7FFFu + ((v.u >> 16) & 1u);
  return (uint16_t)(u >> 16);
}

// ---------------------------------------------------------------- conversions
__global__ void cvt_f32_bf16(const float* __restrict__ in,
                             uint16_t* __restrict__ out, int n) {
  int i = blockIdx.x * blockDim.x + threadIdx.x;
  if (i < n) out[i] = f32_to_bf16_rn(in[i]);
}

// vt[((b*H+h)*D + d)*T + t] = qkv[(b*T+t)*3C + 2C + h*D + d]
__global__ void transpose_v_kernel(const uint16_t* __restrict__ qkv,
                                   uint16_t* __restrict__ vt) {
  size_t idx = (size_t)blockIdx.x * blockDim.x + threadIdx.x; // B*H*D*T
  int t   = (int)(idx & (SEQ - 1));
  size_t rest = idx >> 11;
  int d   = (int)(rest & (HEADDIM - 1));
  int bh  = (int)(rest >> 6);
  int h   = bh & (HEADS - 1);
  int b   = bh >> 4;
  vt[idx] = qkv[((size_t)b * SEQ + t) * (3 * HIDDEN) + 2 * HIDDEN + h * HEADDIM + d];
}

// ---------------------------------------------------------------- WMMA GEMM
// C[M,N] = A[M,K](bf16) * B[K,N](bf16) + bias ; out fp32 or bf16
#define BM 128
#define BN 128
#define BK 64

// Issue a TDM load of a BM x BK bf16 tile (row stride K elems) from gaddr into
// LDS at lds_byte_addr, padding each 128-byte tile row with 16 bytes so the
// LDS image has row stride (BK+8) elements.  D# packing per CDNA5 ISA ch.8.
__device__ __forceinline__ void tdm_load_a_tile(uint64_t gaddr, uint32_t lds_byte_addr,
                                                int K) {
  u32x4 g0;
  g0[0] = 1u;                                          // count=1, user desc
  g0[1] = lds_byte_addr;                               // lds_addr
  g0[2] = (uint32_t)gaddr;                             // global_addr[31:0]
  g0[3] = ((uint32_t)(gaddr >> 32) & 0x01FFFFFFu)      // global_addr[56:32]
        | (2u << 30);                                  // type = 2 (image)
  i32x8 g1;
  g1[0] = (int)((1u << 16)                             // data_size = 2 bytes
              | (1u << 20)                             // pad_enable
              | (4u << 22)                             // pad_interval: 32 DWORDs
              | (3u << 25));                           // pad_amount : 4 DWORDs
  g1[1] = (int)((uint32_t)BK << 16);                   // tensor_dim0 = 64 (lo16 in [31:16])
  g1[2] = (int)((uint32_t)BM << 16);                   // dim0 hi=0 | tensor_dim1 = 128
  g1[3] = (int)((uint32_t)BK << 16);                   // dim1 hi=0 | tile_dim0 = 64
  g1[4] = (int)(uint32_t)BM;                           // tile_dim1 = 128 | tile_dim2 = 0
  g1[5] = (int)(uint32_t)K;                            // tensor_dim0_stride lo32
  g1[6] = 0;                                           // stride0 hi | stride1 lo
  g1[7] = 0;
  i32x4 gz4 = {0, 0, 0, 0};                            // 2-D: groups 2/3 unused
  i32x8 gz8 = {0, 0, 0, 0, 0, 0, 0, 0};
  __builtin_amdgcn_tensor_load_to_lds(g0, g1, gz4, gz4, gz8, 0);
}

__global__ __launch_bounds__(256)
void gemm_bf16_wmma(const uint16_t* __restrict__ A, const uint16_t* __restrict__ B,
                    const float* __restrict__ bias, void* __restrict__ Cout,
                    int M, int N, int K, int out_bf16) {
  __shared__ uint16_t As[BM][BK + 8];   // row-major (A operand), filled by TDM
  __shared__ uint16_t Bs[BN][BK + 8];   // transposed: [col][k] (B operand)

  const int tid  = threadIdx.x;
  const int lane = tid & 31;
  const int wid  = tid >> 5;     // 0..7
  const int wm   = wid >> 2;     // 0..1  (64 rows each)
  const int wn   = wid & 3;      // 0..3  (32 cols each)
  const int half = lane >> 4;
  const int l16  = lane & 15;

  const int mBase = blockIdx.y * BM;
  const int nBase = blockIdx.x * BN;

  const uint32_t ldsA = (uint32_t)(uintptr_t)(&As[0][0]);  // LDS byte offset

  f32x8 acc[4][2];
#pragma unroll
  for (int i = 0; i < 4; ++i)
#pragma unroll
    for (int j = 0; j < 2; ++j) acc[i][j] = f32x8{};

  for (int k0 = 0; k0 < K; k0 += BK) {
    // ---- A tile via Tensor Data Mover (one wave issues; TENSORcnt tracks)
    if (wid == 0) {
      uint64_t gaddr = (uint64_t)(uintptr_t)(A + (size_t)mBase * K + k0);
      tdm_load_a_tile(gaddr, ldsA, K);
    }
    // ---- B tile: 64 x 128, scattered transpose into Bs[col][k]
#pragma unroll
    for (int it = 0; it < 4; ++it) {
      int seg = it * 256 + tid;       // 0..1023
      int kr  = seg >> 4;             // 0..63
      int nc  = (seg & 15) * 8;       // 0..120
      u32x4 v = *reinterpret_cast<const u32x4*>(B + (size_t)(k0 + kr) * N + nBase + nc);
      const uint16_t* e = reinterpret_cast<const uint16_t*>(&v);
#pragma unroll
      for (int q = 0; q < 8; ++q) Bs[nc + q][kr] = e[q];
    }
    if (wid == 0) __builtin_amdgcn_s_wait_tensorcnt(0);
    __syncthreads();

#pragma unroll
    for (int kk = 0; kk < BK; kk += 32) {
      bf16x16 af[4];
#pragma unroll
      for (int i = 0; i < 4; ++i) {
        int row = wm * 64 + i * 16 + l16;
        int kA  = kk + (half ? 8 : 0);
        u32x4 lo = *reinterpret_cast<const u32x4*>(&As[row][kA]);
        u32x4 hi = *reinterpret_cast<const u32x4*>(&As[row][kA + 16]);
        af[i] = frag_from(lo, hi);
      }
      bf16x16 bfr[2];
#pragma unroll
      for (int j = 0; j < 2; ++j) {
        int col = wn * 32 + j * 16 + l16;
        int kB  = kk + (half ? 8 : 0);
        u32x4 lo = *reinterpret_cast<const u32x4*>(&Bs[col][kB]);
        u32x4 hi = *reinterpret_cast<const u32x4*>(&Bs[col][kB + 16]);
        bfr[j] = frag_from(lo, hi);
      }
#pragma unroll
      for (int i = 0; i < 4; ++i)
#pragma unroll
        for (int j = 0; j < 2; ++j)
          acc[i][j] = __builtin_amdgcn_wmma_f32_16x16x32_bf16(
              false, af[i], false, bfr[j], (short)0, acc[i][j], false, false);
    }
    __syncthreads();
  }

  // ---- epilogue: bias + store
#pragma unroll
  for (int i = 0; i < 4; ++i) {
#pragma unroll
    for (int j = 0; j < 2; ++j) {
      int col  = nBase + wn * 32 + j * 16 + l16;
      float bv = bias ? bias[col] : 0.0f;
#pragma unroll
      for (int r = 0; r < 8; ++r) {
        int row = mBase + wm * 64 + i * 16 + r + half * 8;
        float v = acc[i][j][r] + bv;
        if (out_bf16) ((uint16_t*)Cout)[(size_t)row * N + col] = f32_to_bf16_rn(v);
        else          ((float*)Cout)[(size_t)row * N + col]    = v;
      }
    }
  }
}

// ---------------------------------------------------------------- flash attention
// qkv: [B*T][3C] bf16 ; vt: [B*H*D][T] bf16 ; ao: [B*T][C] bf16
__global__ __launch_bounds__(128)
void flash_attn_wmma(const uint16_t* __restrict__ qkv,
                     const uint16_t* __restrict__ vt,
                     uint16_t* __restrict__ ao) {
  __shared__ uint16_t Pst[4][16][40];     // per-wave P staging (C-layout -> A-layout)

  const int lane = threadIdx.x & 31;
  const int wid  = threadIdx.x >> 5;
  const int half = lane >> 4;
  const int l16  = lane & 15;

  const int gw = blockIdx.x * 4 + wid;    // 0..8191
  const int qt = gw & 127;                // T/16 = 128 q-tiles
  const int bh = gw >> 7;                 // 0..63  (= b*H + h)
  const int h  = bh & (HEADS - 1);
  const int b  = bh >> 4;
  const int q0 = qt * 16;

  const size_t rowStride = 3 * HIDDEN;
  const size_t qkvBase   = (size_t)b * SEQ * rowStride;
  const float  scale     = 0.125f;        // 1/sqrt(64)

  // Q fragments (A operand, 2 k-chunks over D=64)
  bf16x16 qa[2];
  {
    const uint16_t* qp = qkv + qkvBase + (size_t)(q0 + l16) * rowStride + h * HEADDIM;
#pragma unroll
    for (int c = 0; c < 2; ++c) {
      int d0 = c * 32 + (half ? 8 : 0);
      int d1 = c * 32 + 16 + (half ? 8 : 0);
      u32x4 lo = *reinterpret_cast<const u32x4*>(qp + d0);
      u32x4 hi = *reinterpret_cast<const u32x4*>(qp + d1);
      qa[c] = frag_from(lo, hi);
    }
  }

  f32x8 o[4];
#pragma unroll
  for (int nt = 0; nt < 4; ++nt) o[nt] = f32x8{};
  float mrow[8], ssum[8];
#pragma unroll
  for (int r = 0; r < 8; ++r) { mrow[r] = -INFINITY; ssum[r] = 0.0f; }

  const int kmax = q0 + 15;
  for (int k0 = 0; k0 <= kmax; k0 += 32) {
    // ---- K B-fragments: S = Q @ K^T  (kdim = d, cols = keys)
    bf16x16 kb[2][2];
#pragma unroll
    for (int nt = 0; nt < 2; ++nt) {
      const uint16_t* kp = qkv + qkvBase + (size_t)(k0 + nt * 16 + l16) * rowStride
                         + HIDDEN + h * HEADDIM;
#pragma unroll
      for (int c = 0; c < 2; ++c) {
        int d0 = c * 32 + (half ? 8 : 0);
        u32x4 lo = *reinterpret_cast<const u32x4*>(kp + d0);
        u32x4 hi = *reinterpret_cast<const u32x4*>(kp + d0 + 16);
        kb[c][nt] = frag_from(lo, hi);
      }
    }

    f32x8 s0 = f32x8{}, s1 = f32x8{};
    s0 = __builtin_amdgcn_wmma_f32_16x16x32_bf16(false, qa[0], false, kb[0][0], (short)0, s0, false, false);
    s0 = __builtin_amdgcn_wmma_f32_16x16x32_bf16(false, qa[1], false, kb[1][0], (short)0, s0, false, false);
    s1 = __builtin_amdgcn_wmma_f32_16x16x32_bf16(false, qa[0], false, kb[0][1], (short)0, s1, false, false);
    s1 = __builtin_amdgcn_wmma_f32_16x16x32_bf16(false, qa[1], false, kb[1][1], (short)0, s1, false, false);

    // ---- scale + causal mask + online softmax
    float p0[8], p1[8], cm[8];
#pragma unroll
    for (int r = 0; r < 8; ++r) {
      int qrow = q0 + r + half * 8;
      float v0 = s0[r] * scale; if (k0 + l16      > qrow) v0 = -3.0e38f;
      float v1 = s1[r] * scale; if (k0 + 16 + l16 > qrow) v1 = -3.0e38f;
      p0[r] = v0; p1[r] = v1;
      cm[r] = fmaxf(v0, v1);
    }
#pragma unroll
    for (int off = 1; off <= 8; off <<= 1)
#pragma unroll
      for (int r = 0; r < 8; ++r)
        cm[r] = fmaxf(cm[r], __shfl_xor(cm[r], off, 32));

    float rs[8];
#pragma unroll
    for (int r = 0; r < 8; ++r) {
      float mnew  = fmaxf(mrow[r], cm[r]);
      float alpha = exp2f((mrow[r] - mnew) * L2E);
      mrow[r] = mnew;
      float e0 = exp2f((p0[r] - mnew) * L2E);
      float e1 = exp2f((p1[r] - mnew) * L2E);
      rs[r] = e0 + e1;
      ssum[r] *= alpha;
#pragma unroll
      for (int nt = 0; nt < 4; ++nt) o[nt][r] *= alpha;
      Pst[wid][r + half * 8][l16]      = f32_to_bf16_rn(e0);
      Pst[wid][r + half * 8][16 + l16] = f32_to_bf16_rn(e1);
    }
#pragma unroll
    for (int off = 1; off <= 8; off <<= 1)
#pragma unroll
      for (int r = 0; r < 8; ++r)
        rs[r] += __shfl_xor(rs[r], off, 32);
#pragma unroll
    for (int r = 0; r < 8; ++r) ssum[r] += rs[r];

    // wave-local LDS round trip: C-layout -> A-layout
    __builtin_amdgcn_wave_barrier();
    asm volatile("s_wait_dscnt 0" ::: "memory");

    bf16x16 pa;
    {
      int kA = half ? 8 : 0;
      u32x4 lo = *reinterpret_cast<const u32x4*>(&Pst[wid][l16][kA]);
      u32x4 hi = *reinterpret_cast<const u32x4*>(&Pst[wid][l16][kA + 16]);
      pa = frag_from(lo, hi);
    }
    __builtin_amdgcn_wave_barrier();

    // ---- V B-fragments from transposed V: elements run over keys (contiguous)
    bf16x16 vb[4];
#pragma unroll
    for (int nt = 0; nt < 4; ++nt) {
      const uint16_t* vp = vt + ((size_t)bh * HEADDIM + nt * 16 + l16) * SEQ
                         + k0 + (half ? 8 : 0);
      u32x4 lo = *reinterpret_cast<const u32x4*>(vp);
      u32x4 hi = *reinterpret_cast<const u32x4*>(vp + 16);
      vb[nt] = frag_from(lo, hi);
    }
#pragma unroll
    for (int nt = 0; nt < 4; ++nt)
      o[nt] = __builtin_amdgcn_wmma_f32_16x16x32_bf16(
          false, pa, false, vb[nt], (short)0, o[nt], false, false);
  }

  // ---- normalize + store attention output (bf16, [B*T][C] with head offset)
#pragma unroll
  for (int r = 0; r < 8; ++r) {
    float inv  = 1.0f / ssum[r];
    int   trow = q0 + r + half * 8;
    size_t base = ((size_t)b * SEQ + trow) * HIDDEN + h * HEADDIM;
#pragma unroll
    for (int nt = 0; nt < 4; ++nt)
      ao[base + nt * 16 + l16] = f32_to_bf16_rn(o[nt][r] * inv);
  }
}

// ---------------------------------------------------------------- launch
extern "C" void kernel_launch(void* const* d_in, const int* in_sizes, int n_in,
                              void* d_out, int out_size, void* d_ws, size_t ws_size,
                              hipStream_t stream) {
  const float* x     = (const float*)d_in[0];
  const float* w_qkv = (const float*)d_in[1];
  const float* b_qkv = (const float*)d_in[2];
  const float* w_out = (const float*)d_in[3];
  const float* b_out = (const float*)d_in[4];
  float* out = (float*)d_out;

  char* ws = (char*)d_ws;
  uint16_t* xb    = (uint16_t*)(ws);                 // 16 MB  x bf16
  uint16_t* wqkvb = (uint16_t*)(ws + 16777216);      //  6 MB  w_qkv bf16
  uint16_t* woutb = (uint16_t*)(ws + 23068672);      //  2 MB  w_out bf16
  uint16_t* qkvb  = (uint16_t*)(ws + 25165824);      // 48 MB  qkv bf16
  uint16_t* vtb   = (uint16_t*)(ws + 75497472);      // 16 MB  V transposed
  uint16_t* aob   = (uint16_t*)(ws + 92274688);      // 16 MB  attn out bf16

  const int nx    = BATCH * SEQ * HIDDEN;            // 8388608
  const int nwqkv = HIDDEN * 3 * HIDDEN;             // 3145728
  const int nwout = HIDDEN * HIDDEN;                 // 1048576

  cvt_f32_bf16<<<(nx    + 255) / 256, 256, 0, stream>>>(x,     xb,    nx);
  cvt_f32_bf16<<<(nwqkv + 255) / 256, 256, 0, stream>>>(w_qkv, wqkvb, nwqkv);
  cvt_f32_bf16<<<(nwout + 255) / 256, 256, 0, stream>>>(w_out, woutb, nwout);

  // qkv = x @ w_qkv + b_qkv : [8192x1024] @ [1024x3072] -> bf16
  gemm_bf16_wmma<<<dim3(3072 / BN, (BATCH * SEQ) / BM), 256, 0, stream>>>(
      xb, wqkvb, b_qkv, qkvb, BATCH * SEQ, 3 * HIDDEN, HIDDEN, 1);

  transpose_v_kernel<<<nx / 256, 256, 0, stream>>>(qkvb, vtb);

  // flash attention: one wave per 16-query tile, 4 waves/block
  flash_attn_wmma<<<(BATCH * HEADS * (SEQ / 16)) / 4, 128, 0, stream>>>(qkvb, vtb, aob);

  // out = ao @ w_out + b_out : [8192x1024] @ [1024x1024] -> fp32
  gemm_bf16_wmma<<<dim3(HIDDEN / BN, (BATCH * SEQ) / BM), 256, 0, stream>>>(
      aob, woutb, b_out, out, BATCH * SEQ, HIDDEN, HIDDEN, 0);
}
